// newNCE_75763223102024
// MI455X (gfx1250) — matrix operand order repaired and checked by reading, not compiled, not run on hardware
//
#include <hip/hip_runtime.h>

// ---------------------------------------------------------------------------
// Contrastive (GRACE-style) loss on MI455X / gfx1250.
// 4 x (16384 x 16384 x 256) Gram row-sum passes via v_wmma_f32_16x16x32_f16,
// exp + rowsum fused into the WMMA tile epilogue. ~550 GFLOP, compute bound.
// Round 3: fix async-load builtin typing: (int4 AS1*, int4 AS3*, imm, imm).
// ---------------------------------------------------------------------------

typedef _Float16 v16h __attribute__((ext_vector_type(16)));
typedef _Float16 v8h  __attribute__((ext_vector_type(8)));
typedef float    v8f  __attribute__((ext_vector_type(8)));

#define N_NODES 16384
#define DIMS    256
#define MTILE   128
#define LDS_ROW 528                         // 256 f16 (512B) + 16B pad: conflict-free 16B reads
#define KEXP    2.8853900817779268f         // (1/tau) * log2(e), tau = 0.5
#define E2      7.3890560989306495f         // exp(1/tau) = exp(2)

#if __has_builtin(__builtin_amdgcn_global_load_async_to_lds_b128)
#define HAVE_ASYNC_LDS 1
#else
#define HAVE_ASYNC_LDS 0
#endif

#if HAVE_ASYNC_LDS
typedef int int4v __attribute__((ext_vector_type(4)));
typedef __attribute__((address_space(1))) int4v *g_b128;   // global int4*
typedef __attribute__((address_space(3))) int4v *l_b128;   // LDS int4*
#endif

static __device__ __forceinline__ void wait_async0() {
#if HAVE_ASYNC_LDS
#if __has_builtin(__builtin_amdgcn_s_wait_asynccnt)
    __builtin_amdgcn_s_wait_asynccnt(0);
#else
    asm volatile("s_wait_asynccnt 0x0" ::: "memory");
#endif
#endif
}

// Stage a 128-row x 512B panel (row stride LDS_ROW in LDS), 256 threads.
static __device__ __forceinline__ void stage_panel(
    char* dst, const _Float16* __restrict__ src, int tid)
{
    for (int idx = tid; idx < MTILE * 32; idx += 256) {
        const int r = idx >> 5, seg = idx & 31;
#if HAVE_ASYNC_LDS
        __builtin_amdgcn_global_load_async_to_lds_b128(
            (g_b128)(void*)(src + (size_t)r * DIMS + seg * 8),
            (l_b128)(void*)(dst + r * LDS_ROW + seg * 16),
            0, 0);
#else
        uint4 v = ((const uint4*)(src + (size_t)r * DIMS))[seg];
        *(uint4*)(dst + r * LDS_ROW + seg * 16) = v;
#endif
    }
}

// ---------------------------------------------------------------------------
// Kernel 1: L2-normalize rows (fp32 math), emit fp16 copies + fp32 diag(an.bn)
// One wave32 per row; 8 rows per 256-thread block.
// ---------------------------------------------------------------------------
__global__ __launch_bounds__(256) void prep_kernel(
    const float* __restrict__ z1, const float* __restrict__ z2,
    _Float16* __restrict__ an, _Float16* __restrict__ bn,
    float* __restrict__ diag)
{
    const int wave = threadIdx.x >> 5;
    const int lane = threadIdx.x & 31;
    const int row  = blockIdx.x * 8 + wave;

    const float4* za = (const float4*)(z1 + (size_t)row * DIMS);
    const float4* zb = (const float4*)(z2 + (size_t)row * DIMS);
    float4 a0 = za[lane * 2], a1 = za[lane * 2 + 1];
    float4 b0 = zb[lane * 2], b1 = zb[lane * 2 + 1];

    float a[8] = {a0.x, a0.y, a0.z, a0.w, a1.x, a1.y, a1.z, a1.w};
    float b[8] = {b0.x, b0.y, b0.z, b0.w, b1.x, b1.y, b1.z, b1.w};

    float sa = 0.f, sb = 0.f, dab = 0.f;
#pragma unroll
    for (int t = 0; t < 8; ++t) {
        sa  += a[t] * a[t];
        sb  += b[t] * b[t];
        dab += a[t] * b[t];
    }
#pragma unroll
    for (int m = 16; m >= 1; m >>= 1) {
        sa  += __shfl_xor(sa,  m, 32);
        sb  += __shfl_xor(sb,  m, 32);
        dab += __shfl_xor(dab, m, 32);
    }
    const float inv_a = 1.0f / fmaxf(__builtin_sqrtf(sa), 1e-12f);
    const float inv_b = 1.0f / fmaxf(__builtin_sqrtf(sb), 1e-12f);

    v8h ha, hb;
#pragma unroll
    for (int t = 0; t < 8; ++t) {
        ha[t] = (_Float16)(a[t] * inv_a);
        hb[t] = (_Float16)(b[t] * inv_b);
    }
    *(v8h*)(an + (size_t)row * DIMS + lane * 8) = ha;
    *(v8h*)(bn + (size_t)row * DIMS + lane * 8) = hb;

    if (lane == 0) diag[row] = dab * inv_a * inv_b;
}

// ---------------------------------------------------------------------------
// Kernel 2: rowsum_i( exp( (X_i . Y_j) / tau ) ) for one of 4 (X,Y) pairs.
// blockIdx.x: 128-row M tile. blockIdx.y: pass 0..3 = aa, ab, bb, ba.
// 8 waves; each wave owns 16 rows and walks all N columns in 128-col chunks.
// Y panel is double-buffered: async copy of chunk jc+1 overlaps WMMA on jc.
// ---------------------------------------------------------------------------
__global__ __launch_bounds__(256) void gram_rowsum_kernel(
    const _Float16* __restrict__ an, const _Float16* __restrict__ bn,
    float* __restrict__ sums)
{
    extern __shared__ char smem[];
    char* xs = smem;                            // X tile:   MTILE * LDS_ROW
    char* yb0 = smem + MTILE * LDS_ROW;         // Y buf 0:  MTILE * LDS_ROW
    char* yb1 = smem + 2 * MTILE * LDS_ROW;     // Y buf 1:  MTILE * LDS_ROW

    const int pass = blockIdx.y;
    const _Float16* X = (pass >= 2) ? bn : an;                   // aa,ab -> an ; bb,ba -> bn
    const _Float16* Y = (pass == 1 || pass == 2) ? bn : an;      // ab,bb -> bn ; aa,ba -> an
    float* out = sums + (size_t)pass * N_NODES;

    const int tid  = threadIdx.x;
    const int wave = tid >> 5;
    const int lane = tid & 31;
    const int half = lane >> 4;                 // lane group within wave
    const int l16  = lane & 15;
    const int m0   = blockIdx.x * MTILE;

    // ---- stage X tile (sync, once) and Y chunk 0 (async) ----
    for (int idx = tid; idx < MTILE * 32; idx += 256) {
        const int r = idx >> 5, seg = idx & 31;
        uint4 v = ((const uint4*)(X + (size_t)(m0 + r) * DIMS))[seg];
        *(uint4*)(xs + r * LDS_ROW + seg * 16) = v;
    }
    stage_panel(yb0, Y, tid);
    wait_async0();
    __syncthreads();

    // ---- load this wave's 8 A fragments (16 rows x K=256) ----
    // 16-bit A 16x32 layout: lanes 0-15 -> K 0..7 / 16..23, lanes 16-31 -> K 8..15 / 24..31
    v16h afrag[8];
    {
        const char* abase = xs + (wave * 16 + l16) * LDS_ROW + half * 16;
#pragma unroll
        for (int k = 0; k < 8; ++k) {
            v8h lo = *(const v8h*)(abase + k * 64);
            v8h hi = *(const v8h*)(abase + k * 64 + 32);
            afrag[k] = __builtin_shufflevector(lo, hi,
                0, 1, 2, 3, 4, 5, 6, 7, 8, 9, 10, 11, 12, 13, 14, 15);
        }
    }

    float rowpart[8];
#pragma unroll
    for (int r = 0; r < 8; ++r) rowpart[r] = 0.f;

    const int NCHUNK = N_NODES / MTILE;
    for (int jc = 0; jc < NCHUNK; ++jc) {
        char* cur = (jc & 1) ? yb1 : yb0;
        // prefetch next chunk into the other buffer while we compute
        if (jc + 1 < NCHUNK)
            stage_panel((jc & 1) ? yb0 : yb1,
                        Y + (size_t)(jc + 1) * MTILE * DIMS, tid);

        // ---- 8 column subtiles of 16; K=256 in 8 WMMA steps each ----
#pragma unroll
        for (int c = 0; c < 8; ++c) {
            // 16-bit B 32x16 layout: lane = column; lanes 0-15 K 0..7 / 16..23, 16-31 K 8..15 / 24..31
            const char* bbase = cur + (c * 16 + l16) * LDS_ROW + half * 16;
            v16h bfrag[8];
#pragma unroll
            for (int k = 0; k < 8; ++k) {
                v8h lo = *(const v8h*)(bbase + k * 64);
                v8h hi = *(const v8h*)(bbase + k * 64 + 32);
                bfrag[k] = __builtin_shufflevector(lo, hi,
                    0, 1, 2, 3, 4, 5, 6, 7, 8, 9, 10, 11, 12, 13, 14, 15);
            }
            v8f acc = {};
#pragma unroll
            for (int k = 0; k < 8; ++k)
                acc = __builtin_amdgcn_wmma_f32_16x16x32_f16(
                    false, afrag[k], false, bfrag[k], (short)0, acc, false, false);

            // epilogue: exp(score/tau) accumulated into per-lane row partials.
            // C layout: VGPR r, lane half -> row (wave*16 + half*8 + r), col = l16.
#pragma unroll
            for (int r = 0; r < 8; ++r)
                rowpart[r] += __builtin_amdgcn_exp2f(acc[r] * KEXP);
        }

        wait_async0();      // next chunk landed in LDS (our own transfers)
        __syncthreads();    // ...and everyone else's; cur is safe to reuse
    }

    // reduce the 16 columns held by the 16 lanes of each half
#pragma unroll
    for (int r = 0; r < 8; ++r) {
        float v = rowpart[r];
#pragma unroll
        for (int m = 8; m >= 1; m >>= 1) v += __shfl_xor(v, m, 32);
        rowpart[r] = v;
    }
    if (l16 == 0) {
        const int base = m0 + wave * 16 + half * 8;
#pragma unroll
        for (int r = 0; r < 8; ++r) out[base + r] = rowpart[r];
    }
}

// ---------------------------------------------------------------------------
// Kernel 3: per-row losses from the 4 sums + diag, reduced to scalar mean.
// ---------------------------------------------------------------------------
__global__ __launch_bounds__(1024) void finalize_kernel(
    const float* __restrict__ diag, const float* __restrict__ sums,
    float* __restrict__ out)
{
    __shared__ float red[32];
    float acc = 0.f;
    for (int i = threadIdx.x; i < N_NODES; i += 1024) {
        const float d    = diag[i];
        const float den1 = sums[i]               + sums[N_NODES + i]     - E2;
        const float den2 = sums[2 * N_NODES + i] + sums[3 * N_NODES + i] - E2;
        acc += 0.5f * ((__logf(den1) - 2.0f * d) + (__logf(den2) - 2.0f * d));
    }
#pragma unroll
    for (int m = 16; m >= 1; m >>= 1) acc += __shfl_xor(acc, m, 32);
    const int wave = threadIdx.x >> 5, lane = threadIdx.x & 31;
    if (lane == 0) red[wave] = acc;
    __syncthreads();
    if (wave == 0) {
        float v = red[lane];
#pragma unroll
        for (int m = 16; m >= 1; m >>= 1) v += __shfl_xor(v, m, 32);
        if (lane == 0) out[0] = v / (float)N_NODES;
    }
}

// ---------------------------------------------------------------------------
extern "C" void kernel_launch(void* const* d_in, const int* in_sizes, int n_in,
                              void* d_out, int out_size, void* d_ws, size_t ws_size,
                              hipStream_t stream)
{
    const float* z1 = (const float*)d_in[0];
    const float* z2 = (const float*)d_in[1];

    char* ws = (char*)d_ws;
    _Float16* an  = (_Float16*)(ws);                                    //  8 MB
    _Float16* bn  = (_Float16*)(ws + (size_t)N_NODES * DIMS * 2);       //  8 MB
    float*    dia = (float*)   (ws + (size_t)N_NODES * DIMS * 4);       // 64 KB
    float*    sms = (float*)   (ws + (size_t)N_NODES * DIMS * 4
                                   + (size_t)N_NODES * 4);              // 256 KB

    prep_kernel<<<N_NODES / 8, 256, 0, stream>>>(z1, z2, an, bn, dia);

    dim3 grid(N_NODES / MTILE, 4, 1);
    size_t shmem = (size_t)3 * MTILE * LDS_ROW;   // X + 2 Y bufs = 198 KB of 320 KB LDS
    gram_rowsum_kernel<<<grid, 256, shmem, stream>>>(an, bn, sms);

    finalize_kernel<<<1, 1024, 0, stream>>>(dia, sms, (float*)d_out);
}
